// GraphAttentionLayer_90228672955072
// MI455X (gfx1250) — matrix operand order repaired and checked by reading, not compiled, not run on hardware
//
#include <hip/hip_runtime.h>
#include <hip/hip_bf16.h>

typedef __attribute__((ext_vector_type(16))) _Float16 v16h;
typedef __attribute__((ext_vector_type(8)))  _Float16 v8h;
typedef __attribute__((ext_vector_type(8)))  float    v8f;

#define LRELU_ALPHA 0.2f
#define MASK_NEG   -30000.0f   // f16-representable stand-in for -9e15

constexpr int N_NODES  = 1024;
constexpr int FDIM     = 64;
constexpr int SLICES   = 64;          // B*T
constexpr int P_STRIDE = N_NODES + 8; // halves; pad breaks 2048B bank stride

// ---------------- Stage 1: Wh^T (f16), s1, s2 ----------------
__global__ void __launch_bounds__(256)
gat_stage1(const float* __restrict__ h,
           const float* __restrict__ W,
           const float* __restrict__ a,
           _Float16* __restrict__ whT,
           float* __restrict__ s1,
           float* __restrict__ s2) {
  __shared__ float Wsh[FDIM][FDIM];   // 16 KB
  __shared__ float a1[FDIM], a2[FDIM];
  const int slice = blockIdx.x;
  const int t = threadIdx.x;
  for (int idx = t; idx < FDIM * FDIM; idx += 256) Wsh[idx / FDIM][idx % FDIM] = W[idx];
  if (t < FDIM) { a1[t] = a[t]; a2[t] = a[FDIM + t]; }
  __syncthreads();
  for (int i = t; i < N_NODES; i += 256) {
    const float* hr = h + ((size_t)slice * N_NODES + i) * FDIM;
    float hreg[FDIM];
#pragma unroll
    for (int k = 0; k < FDIM; ++k) hreg[k] = hr[k];
    float s1acc = 0.f, s2acc = 0.f;
    for (int c = 0; c < FDIM; ++c) {
      float acc = 0.f;
#pragma unroll
      for (int k = 0; k < FDIM; ++k) acc += hreg[k] * Wsh[k][c];
      whT[((size_t)slice * FDIM + c) * N_NODES + i] = (_Float16)acc;
      s1acc += acc * a1[c];
      s2acc += acc * a2[c];
    }
    s1[slice * N_NODES + i] = s1acc;
    s2[slice * N_NODES + i] = s2acc;
  }
}

// ---------------- Stage 2: softmax(adj-masked scores) @ Wh, ELU ----------------
__global__ void __launch_bounds__(256)
gat_stage2(const int* __restrict__ adj,
           const _Float16* __restrict__ whT,
           const float* __restrict__ s1,
           const float* __restrict__ s2,
           float* __restrict__ out) {
  __shared__ __align__(16) _Float16 p[16 * P_STRIDE]; // ~33 KB softmax numerators
  __shared__ float red[16 * FDIM];                    // 4 KB cross-wave reduction
  __shared__ float rowsum[16];

  const int blk   = blockIdx.x;
  const int slice = blk >> 6;
  const int row0  = (blk & 63) << 4;
  const int t     = threadIdx.x;
  const int lane  = t & 31;

  for (int i = t; i < 16 * FDIM; i += 256) red[i] = 0.f;

  // ---- Phase A: masked leaky-relu scores + softmax (adj read exactly once) ----
  const int r = t >> 4;   // row in tile 0..15
  const int g = t & 15;   // thread within row group
  const float s1v = s1[slice * N_NODES + row0 + r];
  const int* adjrow = adj + ((size_t)slice * N_NODES + row0 + r) * N_NODES;
  const float* s2row = s2 + slice * N_NODES;
  float m = MASK_NEG;
  for (int j = g; j < N_NODES; j += 16) {
    float e = s1v + s2row[j];
    e = (e > 0.f) ? e : LRELU_ALPHA * e;
    e = (adjrow[j] > 0) ? e : MASK_NEG;
    p[r * P_STRIDE + j] = (_Float16)e;
    m = fmaxf(m, e);
  }
#pragma unroll
  for (int off = 8; off >= 1; off >>= 1) m = fmaxf(m, __shfl_xor(m, off, 16));
  float sum = 0.f;
  for (int j = g; j < N_NODES; j += 16) {
    float e  = (float)p[r * P_STRIDE + j];
    float pe = __expf(e - m);
    p[r * P_STRIDE + j] = (_Float16)pe;
    sum += pe;
  }
#pragma unroll
  for (int off = 8; off >= 1; off >>= 1) sum += __shfl_xor(sum, off, 16);
  if (g == 0) rowsum[r] = sum;
  __syncthreads();

  // ---- Phase B: (16x1024 f16) x (1024x64 f16) via v_wmma_f32_16x16x32_f16 ----
  const int w = t >> 5;                 // wave 0..7 owns K-chunks w, w+8, w+16, w+24
  v8f acc[4] = {};
  const int arow = lane & 15;
  const int aoff = (lane >= 16) ? 8 : 0;   // A layout: K {0..7,16..23} | {8..15,24..31}
  const int boff = (lane >= 16) ? 8 : 0;   // B layout: K {0..7,16..23} | {8..15,24..31}, col = lane&15
  const _Float16* whTs = whT + (size_t)slice * FDIM * N_NODES;

#pragma unroll
  for (int q = 0; q < 4; ++q) {
    const int kbase = (w + (q << 3)) << 5;
    union { v16h v; v8h h2[2]; } A;
    A.h2[0] = *(const v8h*)&p[arow * P_STRIDE + kbase + aoff];
    A.h2[1] = *(const v8h*)&p[arow * P_STRIDE + kbase + 16 + aoff];
#pragma unroll
    for (int n = 0; n < 4; ++n) {
      const _Float16* bp = whTs + (size_t)((n << 4) + (lane & 15)) * N_NODES + kbase + boff;
      union { v16h v; v8h h2[2]; } Bm;
      Bm.h2[0] = *(const v8h*)(bp);
      Bm.h2[1] = *(const v8h*)(bp + 16);
      acc[n] = __builtin_amdgcn_wmma_f32_16x16x32_f16(false, A.v, false, Bm.v,
                                                      (short)0, acc[n], false, false);
    }
  }

  // ---- combine 8 waves' partial tiles (C/D layout: M = v + 8*(lane>=16), N = lane&15)
  const int mrow = (lane >> 4) << 3;
  const int ncol = lane & 15;
#pragma unroll
  for (int n = 0; n < 4; ++n)
#pragma unroll
    for (int v = 0; v < 8; ++v)
      atomicAdd(&red[(v + mrow) * FDIM + (n << 4) + ncol], acc[n][v]);
  __syncthreads();

  // ---- normalize, ELU, store ----
  float* orow = out + ((size_t)slice * N_NODES + row0) * FDIM;
  for (int i = t; i < 16 * FDIM; i += 256) {
    const int rr = i >> 6;
    float v = red[i] / rowsum[rr];
    v = (v > 0.f) ? v : (__expf(v) - 1.f);
    orow[rr * FDIM + (i & 63)] = v;
  }
}

extern "C" void kernel_launch(void* const* d_in, const int* in_sizes, int n_in,
                              void* d_out, int out_size, void* d_ws, size_t ws_size,
                              hipStream_t stream) {
  (void)in_sizes; (void)n_in; (void)out_size; (void)ws_size;
  const float* h   = (const float*)d_in[0];
  const int*   adj = (const int*)d_in[1];
  const float* W   = (const float*)d_in[2];
  const float* a   = (const float*)d_in[3];
  float* out = (float*)d_out;

  char* ws = (char*)d_ws;
  _Float16* whT = (_Float16*)ws;                                   // 8 MB
  float* s1 = (float*)(ws + (size_t)SLICES * FDIM * N_NODES * 2);  // 256 KB
  float* s2 = s1 + SLICES * N_NODES;                               // 256 KB

  gat_stage1<<<SLICES, 256, 0, stream>>>(h, W, a, whT, s1, s2);
  gat_stage2<<<SLICES * (N_NODES / 16), 256, 0, stream>>>(adj, whT, s1, s2, out);
}
